// MambaModel_87179246174644
// MI455X (gfx1250) — matrix operand order repaired and checked by reading, not compiled, not run on hardware
//
#include <hip/hip_runtime.h>
#include <hip/hip_bf16.h>

// ---- problem constants (match reference) ----
#define B_SZ    16
#define L_SZ    2048
#define DMODEL  256
#define DINNER  512
#define DSTATE  16
#define DTRANK  16
#define DCONV   4
#define NLAYERS 4
#define BL      (B_SZ * L_SZ)   // 32768

typedef __attribute__((ext_vector_type(16))) __bf16          v16bf;
typedef __attribute__((ext_vector_type(8)))  float           v8f;
typedef __attribute__((ext_vector_type(8)))  unsigned short  us8;

static __device__ __forceinline__ unsigned short f2bf(float f) {
    unsigned u = __float_as_uint(f);
    unsigned r = u + 0x7FFFu + ((u >> 16) & 1u);   // round-to-nearest-even
    return (unsigned short)(r >> 16);
}
static __device__ __forceinline__ float bf2f(unsigned short h) {
    return __uint_as_float(((unsigned)h) << 16);
}

// ---------------- weight conversion f32 -> bf16 ----------------
__global__ void k_f32_to_bf16(const float* __restrict__ src,
                              unsigned short* __restrict__ dst, int n) {
    int i = blockIdx.x * blockDim.x + threadIdx.x;
    if (i < n) dst[i] = f2bf(src[i]);
}

// ---------------- input projection: (B,1,L) -> bf16 (B,L,DMODEL) ----------------
__global__ void k_input_proj(const float* __restrict__ x,
                             const float* __restrict__ w,
                             const float* __restrict__ b,
                             unsigned short* __restrict__ U) {
    int i  = blockIdx.x * blockDim.x + threadIdx.x;   // BL*DMODEL threads, exact
    int d  = i & (DMODEL - 1);
    int bl = i >> 8;
    U[i] = f2bf(x[bl] * w[d] + b[d]);
}

// ---------------- generic bf16 WMMA GEMM: C[M,N] = A[M,K_] * W[N,K_]^T ----------------
// Compile-time K_/LDA/LDC/ACT/HASBIAS -> no runtime guards in the hot loop.
// block = 256 threads = 8 waves; block tile 128(M) x 16(N); K step 32.
// Requires: LDA >= 32 and LDA >= K_ (true for all call sites); N tile exact.
// For K_ < 32 the W stage zero-fills k >= K_, so A may stage in-row garbage there.
template <int K_, int LDA, int LDC, int ACT, bool HASBIAS>
__global__ __launch_bounds__(256)
void k_gemm_bf16(const unsigned short* __restrict__ A,
                 const unsigned short* __restrict__ W,   // N x K_ row-major
                 unsigned short* __restrict__ C,
                 const float* __restrict__ bias) {
    // A tile 128x32, row stride 40 ushorts (80B: 16B-aligned, staggers banks)
    __shared__ __align__(16) unsigned short ldsA[128 * 40];
    // B tile already in per-lane fragment order: [lane][16]
    __shared__ __align__(16) unsigned short ldsBf[32 * 16];

    const int t    = threadIdx.x;
    const int wv   = t >> 5;
    const int lane = t & 31;
    const int half = lane >> 4;
    const int l16  = lane & 15;
    const int m0   = blockIdx.x * 128;
    const int n0   = blockIdx.y * 16;

    union CF { v8f v; float f[8]; } acc;
#pragma unroll
    for (int i = 0; i < 8; ++i) acc.f[i] = 0.f;

    for (int k0 = 0; k0 < K_; k0 += 32) {
        // ---- stage A tile: 512 x 16B chunks, 2 per thread, unguarded b128 ----
#pragma unroll
        for (int j = 0; j < 2; ++j) {
            int idx = t + 256 * j;
            int r   = idx >> 2;          // 0..127
            int c   = idx & 3;           // 4 chunks of 8 ushorts
            uint4 v = *(const uint4*)(A + (size_t)(m0 + r) * LDA + k0 + 8 * c);
            *(uint4*)(ldsA + r * 40 + 8 * c) = v;
        }
        // ---- stage W tile into fragment order, branch-free: 1 uint per thread ----
        // ldsBf[row][ii] = W[n0 + (row&15)][k0 + (row>>4)*16 + ii], row = kh*16+n
        {
            int row = t >> 3;            // 0..31
            int ii  = (t & 7) << 1;      // even element within 16
            int kk  = k0 + ((row >> 4) << 4) + ii;
            unsigned u = 0;
            if (K_ % 32 == 0 || kk < K_)                 // folds away when K_%32==0
                u = *(const unsigned*)(W + (size_t)(n0 + (row & 15)) * K_ + kk);
            *(unsigned*)(ldsBf + row * 16 + ii) = u;
        }
        __syncthreads();

        // ---- fragments: contiguous 16B LDS reads ----
        union VF { v16bf v; us8 h[2]; } fa, fb;
        const unsigned short* arow = ldsA + ((wv << 4) + l16) * 40;
        fa.h[0] = *(const us8*)(arow + 16 * half);       // K = [16h, 16h+8)
        fa.h[1] = *(const us8*)(arow + 16 * half + 8);   // K = [16h+8, 16h+16)
        const unsigned short* brow = ldsBf + lane * 16;
        fb.h[0] = *(const us8*)(brow);
        fb.h[1] = *(const us8*)(brow + 8);

        acc.v = __builtin_amdgcn_wmma_f32_16x16x32_bf16(
            false, fa.v, false, fb.v, (short)0, acc.v, false, false);

        // prefetch next A tile (global_prefetch_b8)
        if (k0 + 32 < K_)
            __builtin_prefetch(A + (size_t)(m0 + (t >> 1)) * LDA + k0 + 32 + ((t & 1) << 4), 0, 1);
        __syncthreads();
    }

    // ---- epilogue: VGPR r -> row (r + 8*half), col l16 ----
#pragma unroll
    for (int r = 0; r < 8; ++r) {
        int m = m0 + (wv << 4) + r + (half << 3);
        int n = n0 + l16;
        float v = acc.f[r];
        if (HASBIAS) v += bias[n];
        if (ACT == 1) v = (v > 20.f) ? v : log1pf(__expf(v));  // softplus
        C[(size_t)m * LDC + n] = f2bf(v);
    }
}

// ---------------- causal depthwise conv (k=4) + SiLU ----------------
__global__ void k_conv_silu(const unsigned short* __restrict__ XZ,   // (BL, 1024), first 512 = xh
                            const float* __restrict__ cw,            // (DINNER, DCONV)
                            const float* __restrict__ cb,            // (DINNER)
                            unsigned short* __restrict__ XC) {       // (BL, 512)
    int i  = blockIdx.x * blockDim.x + threadIdx.x;  // BL*DINNER, exact
    int e  = i & (DINNER - 1);
    int bl = i >> 9;
    int l  = bl & (L_SZ - 1);
    float s = cb[e];
#pragma unroll
    for (int k = 0; k < DCONV; ++k) {
        int ls = l - (DCONV - 1) + k;
        if (ls >= 0)
            s += bf2f(XZ[(size_t)(bl - (DCONV - 1) + k) * (2 * DINNER) + e]) * cw[e * DCONV + k];
    }
    float sig = 1.f / (1.f + __expf(-s));
    XC[i] = f2bf(s * sig);
}

// ---------------- selective scan, fused with D-skip and SiLU(z) gate ----------------
// grid = B, block = DINNER threads; one thread per channel; h-state in VGPRs.
// B/C vectors staged per 32-step chunk into LDS as fp32, read as float4 (ds_load_b128).
__global__ __launch_bounds__(DINNER)
void k_scan(const unsigned short* __restrict__ XC,    // (BL, 512)
            const unsigned short* __restrict__ DT,    // (BL, 512)
            const unsigned short* __restrict__ DBL,   // (BL, 48): [16:32)=B, [32:48)=C
            const unsigned short* __restrict__ XZ,    // (BL, 1024): z = cols [512:1024)
            const float* __restrict__ A_log,          // (DINNER, DSTATE)
            const float* __restrict__ Dp,             // (DINNER)
            unsigned short* __restrict__ Y2) {        // (BL, 512)
    const int b = blockIdx.x;
    const int e = threadIdx.x;
    __shared__ __align__(16) float bcf[32][32];  // 32 steps x (B[16] | C[16]) as fp32

    float a[DSTATE], h[DSTATE];
#pragma unroll
    for (int n = 0; n < DSTATE; ++n) {
        a[n] = -__expf(A_log[e * DSTATE + n]);
        h[n] = 0.f;
    }
    const float de = Dp[e];

    for (int l0 = 0; l0 < L_SZ; l0 += 32) {
        __syncthreads();   // protect previous chunk's LDS reads
        {   // cooperative load+convert of B/C for 32 timesteps: 512 uints, 1/thread
            int lr = e >> 4;
            int pr = e & 15;
            const unsigned short* p = DBL + (size_t)(b * L_SZ + l0 + lr) * 48 + DTRANK + (pr << 1);
            unsigned u = *(const unsigned*)p;
            bcf[lr][(pr << 1)]     = bf2f((unsigned short)u);
            bcf[lr][(pr << 1) + 1] = bf2f((unsigned short)(u >> 16));
        }
        __syncthreads();

        for (int li = 0; li < 32; ++li) {
            int bl = b * L_SZ + l0 + li;
            float xv  = bf2f(XC[(size_t)bl * DINNER + e]);
            float dtv = bf2f(DT[(size_t)bl * DINNER + e]);
            float zv  = bf2f(XZ[(size_t)bl * (2 * DINNER) + DINNER + e]);
            float dx  = dtv * xv;

            const float4* row = (const float4*)&bcf[li][0];
            float4 Bq[4];
            Bq[0] = row[0]; Bq[1] = row[1];   // B[0..7]
            Bq[2] = row[2]; Bq[3] = row[3];   // B[8..15]
            const float* Bv = (const float*)&Bq[0];
            const float* Cv = (const float*)&Bq[2];

            float y = 0.f;
#pragma unroll
            for (int n = 0; n < DSTATE; ++n) {
                float hv = h[n] * __expf(dtv * a[n]) + dx * Bv[n];
                h[n] = hv;
                y += hv * Cv[n];
            }
            float sig = 1.f / (1.f + __expf(-zv));
            Y2[(size_t)bl * DINNER + e] = f2bf((y + xv * de) * (zv * sig));
        }
    }
}

// ---------------- mean over L -> fp32 output (B, DMODEL) ----------------
__global__ void k_mean(const unsigned short* __restrict__ U, float* __restrict__ out) {
    int i = blockIdx.x * blockDim.x + threadIdx.x;  // B*DMODEL, exact
    int d = i & (DMODEL - 1);
    int b = i >> 8;
    float s = 0.f;
    for (int l = 0; l < L_SZ; ++l)
        s += bf2f(U[(size_t)(b * L_SZ + l) * DMODEL + d]);
    out[i] = s * (1.f / (float)L_SZ);
}

// ======================================================================
extern "C" void kernel_launch(void* const* d_in, const int* in_sizes, int n_in,
                              void* d_out, int out_size, void* d_ws, size_t ws_size,
                              hipStream_t stream) {
    (void)in_sizes; (void)n_in; (void)out_size; (void)ws_size;

    const float* x      = (const float*)d_in[0];
    const float* ipw    = (const float*)d_in[1];
    const float* ipb    = (const float*)d_in[2];
    const float* in_w   = (const float*)d_in[3];
    const float* conv_w = (const float*)d_in[4];
    const float* conv_b = (const float*)d_in[5];
    const float* x_w    = (const float*)d_in[6];
    const float* dt_w   = (const float*)d_in[7];
    const float* dt_b   = (const float*)d_in[8];
    const float* A_log  = (const float*)d_in[9];
    const float* Dp     = (const float*)d_in[10];
    const float* out_w  = (const float*)d_in[11];

    char* ws = (char*)d_ws;
    size_t off = 0;
    auto alloc = [&](size_t bytes) -> void* {
        void* p = ws + off;
        off += (bytes + 255) & ~(size_t)255;
        return p;
    };
    // bf16 weight copies
    unsigned short* wb_in  = (unsigned short*)alloc((size_t)NLAYERS * 2 * DINNER * DMODEL * 2);
    unsigned short* wb_x   = (unsigned short*)alloc((size_t)NLAYERS * (DTRANK + 2 * DSTATE) * DINNER * 2);
    unsigned short* wb_dt  = (unsigned short*)alloc((size_t)NLAYERS * DINNER * DTRANK * 2);
    unsigned short* wb_out = (unsigned short*)alloc((size_t)NLAYERS * DMODEL * DINNER * 2);
    // activations (bf16)
    unsigned short* U0   = (unsigned short*)alloc((size_t)BL * DMODEL * 2);
    unsigned short* U1   = (unsigned short*)alloc((size_t)BL * DMODEL * 2);
    unsigned short* XZ   = (unsigned short*)alloc((size_t)BL * 2 * DINNER * 2);
    unsigned short* XC   = (unsigned short*)alloc((size_t)BL * DINNER * 2);
    unsigned short* DBLb = (unsigned short*)alloc((size_t)BL * (DTRANK + 2 * DSTATE) * 2);
    unsigned short* DTb  = (unsigned short*)alloc((size_t)BL * DINNER * 2);
    unsigned short* Y2b  = (unsigned short*)alloc((size_t)BL * DINNER * 2);

    auto cvt = [&](const float* s, unsigned short* d, int n) {
        k_f32_to_bf16<<<(n + 255) / 256, 256, 0, stream>>>(s, d, n);
    };
    cvt(in_w,  wb_in,  NLAYERS * 2 * DINNER * DMODEL);
    cvt(x_w,   wb_x,   NLAYERS * (DTRANK + 2 * DSTATE) * DINNER);
    cvt(dt_w,  wb_dt,  NLAYERS * DINNER * DTRANK);
    cvt(out_w, wb_out, NLAYERS * DMODEL * DINNER);

    k_input_proj<<<(BL * DMODEL) / 256, 256, 0, stream>>>(x, ipw, ipb, U0);

    unsigned short* cur = U0;
    unsigned short* nxt = U1;
    for (int i = 0; i < NLAYERS; ++i) {
        // xz = u @ Wi^T : (BL,256) x (1024,256)^T
        k_gemm_bf16<DMODEL, DMODEL, 2 * DINNER, 0, false>
            <<<dim3(BL / 128, (2 * DINNER) / 16), 256, 0, stream>>>(
                cur, wb_in + (size_t)i * 2 * DINNER * DMODEL, XZ, nullptr);
        // causal depthwise conv + silu
        k_conv_silu<<<(BL * DINNER) / 256, 256, 0, stream>>>(
            XZ, conv_w + (size_t)i * DINNER * DCONV, conv_b + (size_t)i * DINNER, XC);
        // dbl = xh @ Wx^T : (BL,512) x (48,512)^T
        k_gemm_bf16<DINNER, DINNER, DTRANK + 2 * DSTATE, 0, false>
            <<<dim3(BL / 128, (DTRANK + 2 * DSTATE) / 16), 256, 0, stream>>>(
                XC, wb_x + (size_t)i * (DTRANK + 2 * DSTATE) * DINNER, DBLb, nullptr);
        // dt = softplus(dbl[:, :16] @ Wdt^T + bdt) : K=16 (W zero-padded to 32)
        k_gemm_bf16<DTRANK, DTRANK + 2 * DSTATE, DINNER, 1, true>
            <<<dim3(BL / 128, DINNER / 16), 256, 0, stream>>>(
                DBLb, wb_dt + (size_t)i * DINNER * DTRANK, DTb, dt_b + (size_t)i * DINNER);
        // selective scan + D-skip + silu(z) gate (fused)
        k_scan<<<B_SZ, DINNER, 0, stream>>>(
            XC, DTb, DBLb, XZ,
            A_log + (size_t)i * DINNER * DSTATE, Dp + (size_t)i * DINNER, Y2b);
        // u_next = y2 @ Wo^T : (BL,512) x (256,512)^T
        k_gemm_bf16<DINNER, DINNER, DMODEL, 0, false>
            <<<dim3(BL / 128, DMODEL / 16), 256, 0, stream>>>(
                Y2b, wb_out + (size_t)i * DMODEL * DINNER, nxt, nullptr);
        unsigned short* tmp = cur; cur = nxt; nxt = tmp;
    }

    k_mean<<<(B_SZ * DMODEL) / 256, 256, 0, stream>>>(cur, (float*)d_out);
}
